// VectorQuantizer_76124000354526
// MI455X (gfx1250) — matrix operand order repaired and checked by reading, not compiled, not run on hardware
//
#include <hip/hip_runtime.h>
#include <math.h>

typedef __attribute__((ext_vector_type(16))) _Float16 v16h;
typedef __attribute__((ext_vector_type(8)))  _Float16 v8h;
typedef __attribute__((ext_vector_type(8)))  float    v8f;

#define N_TOKENS   131072      // 32 * 64 * 64
#define DIM        64
#define KCODES     1024
#define TOK_PER_BLK 128
#define CHUNK      256         // codebook codes staged per LDS chunk
#define NCHUNKS    (KCODES / CHUNK)
#define OUT_Q      8388608     // 32*64*64*64 quantized elements

union Frag16 { v16h v; v8h h[2]; };

// ---------------- kernel 0: zero the accumulators in workspace ----------------
__global__ void vq_zero(unsigned* counts, float* sse) {
    int t = threadIdx.x;
    if (t < KCODES) counts[t] = 0u;
    if (t == 0) *sse = 0.0f;
}

// ---------------- kernel 1: WMMA distance + argmin ----------------
// grid: N_TOKENS/128 blocks of 256 threads (8 waves). Each wave: 16 tokens.
__global__ void __launch_bounds__(256)
vq_argmin(const float* __restrict__ x, const float* __restrict__ cb,
          int* __restrict__ indices, unsigned* __restrict__ counts) {
    __shared__ _Float16 xs[TOK_PER_BLK * DIM];  // 16 KB
    __shared__ _Float16 cs[CHUNK * DIM];        // 32 KB
    __shared__ float    cn2[CHUNK];             // 1 KB

    const int tid   = threadIdx.x;
    const int lane  = tid & 31;
    const int wave  = tid >> 5;
    const int m     = lane & 15;   // row within 16 (A) / col within 16 (B,C,D)
    const int kh    = lane >> 4;   // lane-half selects K sub-range

    const int token0 = blockIdx.x * TOK_PER_BLK;
    const int bimg   = token0 >> 12;          // 4096 tokens per image, 128 | 4096
    const int wh0    = token0 & 4095;
    const float* xb  = x + (size_t)bimg * DIM * 4096 + wh0;

    // ---- stage x tile: [128 tokens][64 ch], coalesced over tokens ----
    #pragma unroll 4
    for (int i = tid; i < TOK_PER_BLK * DIM; i += 256) {
        int t = i & (TOK_PER_BLK - 1);
        int c = i >> 7;
        xs[t * DIM + c] = (_Float16)xb[(size_t)c * 4096 + t];
    }
    __syncthreads();

    // ---- build A fragments once (contiguous K order, matching B) ----
    // element j (0..7):  K = kh*16 + j       -> v8h group [2*kh]
    // element j (8..15): K = kh*16 + j       -> v8h group [2*kh+1]
    const v8h* xrow = (const v8h*)&xs[(wave * 16 + m) * DIM];
    Frag16 a0, a1;
    a0.h[0] = xrow[2 * kh];     a0.h[1] = xrow[2 * kh + 1];   // K 0..31
    a1.h[0] = xrow[2 * kh + 4]; a1.h[1] = xrow[2 * kh + 5];   // K 32..63

    float bestd[8];
    int   bestidx[8];
    #pragma unroll
    for (int r = 0; r < 8; ++r) { bestd[r] = 3.4e38f; bestidx[r] = 0; }

    for (int chunk = 0; chunk < NCHUNKS; ++chunk) {
        __syncthreads();  // previous chunk's LDS reads done
        // stage codebook chunk as f16 (coalesced) + per-code squared norm (f32)
        const float* cbase = cb + (size_t)chunk * CHUNK * DIM;
        #pragma unroll 4
        for (int i = tid; i < CHUNK * DIM; i += 256) {
            cs[i] = (_Float16)cbase[i];
        }
        {
            const float* cr = cbase + (size_t)tid * DIM;
            float s = 0.0f;
            #pragma unroll
            for (int c = 0; c < DIM; ++c) { float v = cr[c]; s += v * v; }
            cn2[tid] = s;
        }
        __syncthreads();

        // 16 code-tiles of 16 codes each
        for (int t = 0; t < 16; ++t) {
            int codeL = t * 16 + m;                 // this lane's column/code
            // B in contiguous K order, matching A: lane-half kh holds 16 contiguous K values
            const v16h* crow = (const v16h*)&cs[codeL * DIM];
            v16h b0 = crow[kh];                     // K 0..31 (half kh*16..)
            v16h b1 = crow[kh + 2];                 // K 32..63

            v8f acc = {};
            acc = __builtin_amdgcn_wmma_f32_16x16x32_f16(
                      false, a0.v, false, b0, (short)0, acc, false, false);
            acc = __builtin_amdgcn_wmma_f32_16x16x32_f16(
                      false, a1.v, false, b1, (short)0, acc, false, false);

            float cd = cn2[codeL];
            int gidx = chunk * CHUNK + codeL;
            #pragma unroll
            for (int r = 0; r < 8; ++r) {
                float d = cd - 2.0f * acc[r];       // ||x||^2 constant: dropped
                if (d < bestd[r]) { bestd[r] = d; bestidx[r] = gidx; }
            }
        }
    }

    // ---- cross-lane argmin over the 16 lanes sharing each token row ----
    #pragma unroll
    for (int off = 8; off >= 1; off >>= 1) {
        #pragma unroll
        for (int r = 0; r < 8; ++r) {
            float od = __shfl_xor(bestd[r], off, 32);
            int   oi = __shfl_xor(bestidx[r], off, 32);
            if (od < bestd[r] || (od == bestd[r] && oi < bestidx[r])) {
                bestd[r] = od; bestidx[r] = oi;
            }
        }
    }
    if (m == 0) {
        #pragma unroll
        for (int r = 0; r < 8; ++r) {
            int tok = token0 + wave * 16 + r + 8 * kh;  // C/D row = r + 8*(lane>>4)
            indices[tok] = bestidx[r];
            atomicAdd(&counts[bestidx[r]], 1u);
        }
    }
}

// ---------------- kernel 2: quantized output (NCHW) + SSE ----------------
__global__ void __launch_bounds__(256)
vq_write(const float* __restrict__ x, const float* __restrict__ cb,
         const int* __restrict__ indices, float* __restrict__ out,
         float* __restrict__ sse) {
    size_t o  = (size_t)blockIdx.x * 256 + threadIdx.x;   // OUT_Q total, exact
    int wh    = (int)(o & 4095);
    int c     = (int)((o >> 12) & 63);
    int bimg  = (int)(o >> 18);
    int tok   = (bimg << 12) | wh;
    int idx   = indices[tok];
    float q   = cb[(size_t)idx * DIM + c];
    out[o]    = q;
    float dfe = q - x[o];
    float s   = dfe * dfe;

    // wave reduce, then block reduce, one atomic per block
    #pragma unroll
    for (int off = 16; off >= 1; off >>= 1) s += __shfl_down(s, off, 32);
    __shared__ float wsum[8];
    int lane = threadIdx.x & 31, wave = threadIdx.x >> 5;
    if (lane == 0) wsum[wave] = s;
    __syncthreads();
    if (threadIdx.x == 0) {
        float t = 0.0f;
        #pragma unroll
        for (int w = 0; w < 8; ++w) t += wsum[w];
        atomicAdd(sse, t);
    }
}

// ---------------- kernel 3: perplexity + scalar losses ----------------
__global__ void __launch_bounds__(1024)
vq_final(const unsigned* __restrict__ counts, const float* __restrict__ sse,
         float* __restrict__ out_tail) {
    int t = threadIdx.x;
    float cnt = (float)counts[t];
    const float logN = logf((float)N_TOKENS);
    float contrib = (cnt > 0.0f)
                  ? (cnt / (float)N_TOKENS) * (logf(cnt) - logN) : 0.0f;
    #pragma unroll
    for (int off = 16; off >= 1; off >>= 1) contrib += __shfl_down(contrib, off, 32);
    __shared__ float wsum[32];
    int lane = t & 31, wave = t >> 5;
    if (lane == 0) wsum[wave] = contrib;
    __syncthreads();
    if (t == 0) {
        float tot = 0.0f;
        #pragma unroll
        for (int w = 0; w < 32; ++w) tot += wsum[w];
        float entropy = -tot;
        float ppl_loss = 1.0f / expf(entropy);
        float mse = *sse / (float)OUT_Q;
        out_tail[0] = mse;                               // codebook_loss
        out_tail[1] = mse;                               // commitment_loss
        out_tail[2] = ppl_loss;                          // perplexity_loss
        out_tail[3] = mse + 0.25f * mse + 0.25f * ppl_loss;  // total loss
    }
}

extern "C" void kernel_launch(void* const* d_in, const int* in_sizes, int n_in,
                              void* d_out, int out_size, void* d_ws, size_t ws_size,
                              hipStream_t stream) {
    const float* x  = (const float*)d_in[0];   // [32,64,64,64]
    const float* cb = (const float*)d_in[1];   // [1024,64]
    float* out      = (float*)d_out;           // quantized + 4 scalars

    // workspace layout: counts[1024] u32 | sse[1] f32 | pad | indices[131072] i32
    unsigned* counts  = (unsigned*)d_ws;
    float*    sse     = (float*)d_ws + KCODES;
    int*      indices = (int*)d_ws + KCODES + 32;

    vq_zero<<<1, 1024, 0, stream>>>(counts, sse);
    vq_argmin<<<N_TOKENS / TOK_PER_BLK, 256, 0, stream>>>(x, cb, indices, counts);
    vq_write<<<OUT_Q / 256, 256, 0, stream>>>(x, cb, indices, out, sse);
    vq_final<<<1, 1024, 0, stream>>>(counts, sse, out + OUT_Q);
}
